// MultiScaleOneRec_26405458936364
// MI455X (gfx1250) — matrix operand order repaired and checked by reading, not compiled, not run on hardware
//
#include <hip/hip_runtime.h>
#include <hip/hip_bf16.h>
#include <math.h>

typedef long long i64;
typedef __bf16 bf16;
typedef __attribute__((ext_vector_type(16))) __bf16 v16bf;
typedef __attribute__((ext_vector_type(8)))  __bf16 v8bf;
typedef __attribute__((ext_vector_type(8)))  float   v8f;
typedef __attribute__((ext_vector_type(4))) unsigned u32x4;
typedef __attribute__((ext_vector_type(8))) int      i32x8;
typedef __attribute__((ext_vector_type(4))) int      i32x4;

#ifndef __has_builtin
#define __has_builtin(x) 0
#endif
#if __has_builtin(__builtin_amdgcn_tensor_load_to_lds) && \
    __has_builtin(__builtin_amdgcn_tensor_store_from_lds) && \
    __has_builtin(__builtin_amdgcn_s_wait_tensorcnt)
#define HAVE_TDM 1
#else
#define HAVE_TDM 0
#endif

namespace {
constexpr int Bc = 8, Sc = 512, Dc = 512, Kc = 1024, LCc = 4, NL = 2;
constexpr int Hc = 8, DKc = 64, DFFc = 2048, Ec = 8;
constexpr int Rw = Bc * Sc;            // 4096 token rows
constexpr float NEGc = -1.0e9f;
constexpr int TDM_CHUNK = 16384;       // floats per TDM chunk (64 KB LDS bounce)
}

__device__ __forceinline__ bf16 f2bf(float f) {
  union { float f; unsigned u; } x; x.f = f;
  unsigned r = (x.u + 0x7FFFu + ((x.u >> 16) & 1u)) >> 16;   // RNE
  union { unsigned short s; bf16 b; } y; y.s = (unsigned short)r;
  return y.b;
}

// ---------------------------------------------------------------------------
// Generic strided batched GEMM: D[z][m][n] = alpha * sum_k A(m,k)*B(k,n) (+R)(+ReLU)
// z = zb*zdiv + zh with separate strides covers per-(b,h) attention einsums.
// ---------------------------------------------------------------------------
struct GemmP {
  const float* A;  i64 am, ak, ab, ah;
  const float* Bm; i64 bk, bn, bb, bh;
  const float* Rr; i64 rm, rb, rh;
  float* Dd;       i64 dm, db, dh;
  int M, N, K, zdiv, flags;   // flags: 1=ReLU, 2=add residual R
  float alpha;
};

#define BM 128
#define BN 128
#define BKK 32

__global__ __launch_bounds__(256) void k_gemm(GemmP p) {
  __shared__ bf16 As[BM * BKK];   // [m][k]
  __shared__ bf16 Bs[BN * BKK];   // [n][k]  (B staged transposed)

  const int tid  = threadIdx.x;
  const int lane = tid & 31;
  const int wave = tid >> 5;
  const int wm   = wave & 3;      // 4 row groups * 32 rows
  const int wn   = wave >> 2;     // 2 col groups * 64 cols
  const int lr   = lane & 15;
  const int hi16 = lane >> 4;     // 0: lanes 0-15, 1: lanes 16-31

  const int m0 = blockIdx.y * BM;
  const int n0 = blockIdx.x * BN;
  const int z  = blockIdx.z;
  const int zb = z / p.zdiv, zh = z % p.zdiv;

  const float* Ab = p.A  + (i64)zb * p.ab + (i64)zh * p.ah;
  const float* Bb = p.Bm + (i64)zb * p.bb + (i64)zh * p.bh;
  float*       Db = p.Dd + (i64)zb * p.db + (i64)zh * p.dh;
  const float* Rb = (p.flags & 2) ? (p.Rr + (i64)zb * p.rb + (i64)zh * p.rh) : nullptr;

  const bool fullK  = (p.K % BKK) == 0;
  const bool fastA  = (p.ak == 1) && (m0 + BM <= p.M) && fullK;  // k-contiguous A
  const bool fastBn = (p.bn == 1) && (n0 + BN <= p.N) && fullK;  // n-contiguous B
  const bool fastBk = (p.bk == 1) && (n0 + BN <= p.N) && fullK;  // k-contiguous B^T

  v8f acc[2][4];
#pragma unroll
  for (int i = 0; i < 2; i++)
#pragma unroll
    for (int j = 0; j < 4; j++)
#pragma unroll
      for (int r = 0; r < 8; r++) acc[i][j][r] = 0.0f;

  for (int k0 = 0; k0 < p.K; k0 += BKK) {
    // ---- stage A (128x32) into LDS ----
    if (fastA) {
#pragma unroll
      for (int f = 0; f < 4; f++) {
        int lin4 = f * 256 + tid;
        int m  = lin4 >> 3;
        int kk = (lin4 & 7) * 4;
        const float4 v = *(const float4*)(Ab + (i64)(m0 + m) * p.am + (k0 + kk));
        bf16* d = As + m * BKK + kk;
        d[0] = f2bf(v.x); d[1] = f2bf(v.y); d[2] = f2bf(v.z); d[3] = f2bf(v.w);
      }
    } else {
#pragma unroll 4
      for (int it = 0; it < 16; it++) {
        int lin = it * 256 + tid;
        int m = lin >> 5, kk = lin & 31;
        float v = 0.0f;
        if (m0 + m < p.M && k0 + kk < p.K)
          v = Ab[(i64)(m0 + m) * p.am + (i64)(k0 + kk) * p.ak];
        As[m * BKK + kk] = f2bf(v);
      }
    }
    // ---- stage B (32x128, transposed to [n][k]) into LDS ----
    if (fastBn) {
#pragma unroll
      for (int f = 0; f < 4; f++) {
        int lin4 = f * 256 + tid;
        int kk = lin4 >> 5;
        int n  = (lin4 & 31) * 4;
        const float4 v = *(const float4*)(Bb + (i64)(k0 + kk) * p.bk + (n0 + n));
        Bs[(n + 0) * BKK + kk] = f2bf(v.x);
        Bs[(n + 1) * BKK + kk] = f2bf(v.y);
        Bs[(n + 2) * BKK + kk] = f2bf(v.z);
        Bs[(n + 3) * BKK + kk] = f2bf(v.w);
      }
    } else if (fastBk) {
#pragma unroll
      for (int f = 0; f < 4; f++) {
        int lin4 = f * 256 + tid;
        int n  = lin4 >> 3;
        int kk = (lin4 & 7) * 4;
        const float4 v = *(const float4*)(Bb + (i64)(n0 + n) * p.bn + (k0 + kk));
        bf16* d = Bs + n * BKK + kk;
        d[0] = f2bf(v.x); d[1] = f2bf(v.y); d[2] = f2bf(v.z); d[3] = f2bf(v.w);
      }
    } else {
#pragma unroll 4
      for (int it = 0; it < 16; it++) {
        int lin = it * 256 + tid;
        int kk = lin >> 7, n = lin & 127;
        float v = 0.0f;
        if (k0 + kk < p.K && n0 + n < p.N)
          v = Bb[(i64)(k0 + kk) * p.bk + (i64)(n0 + n) * p.bn];
        Bs[n * BKK + kk] = f2bf(v);
      }
    }
    __syncthreads();

    // speculative prefetch of next K-tile -> global_prefetch_b8
    if (k0 + BKK < p.K) {
      __builtin_prefetch(Ab + (i64)(m0 + (tid >> 1)) * p.am + (i64)(k0 + BKK) * p.ak, 0, 0);
      __builtin_prefetch(Bb + (i64)(k0 + BKK + (tid >> 7)) * p.bk + (i64)(n0 + (tid & 127)) * p.bn, 0, 0);
    }

    // ---- fragments per documented 16-bit WMMA VGPR layouts ----
    v16bf afrag[2], bfrag[4];
#pragma unroll
    for (int i = 0; i < 2; i++) {
      int row = wm * 32 + i * 16 + lr;
      int kb  = hi16 * 8;                       // K 0-7 / 8-15, then +16
      v8bf lo = *(const v8bf*)(As + row * BKK + kb);
      v8bf hi = *(const v8bf*)(As + row * BKK + kb + 16);
      v16bf f;
#pragma unroll
      for (int t = 0; t < 8; t++) { f[t] = lo[t]; f[t + 8] = hi[t]; }
      afrag[i] = f;
    }
#pragma unroll
    for (int j = 0; j < 4; j++) {
      int col = wn * 64 + j * 16 + lr;
      int kb  = hi16 * 8;                       // K 0-7 / 8-15, then +16
      v8bf lo = *(const v8bf*)(Bs + col * BKK + kb);
      v8bf hi = *(const v8bf*)(Bs + col * BKK + kb + 16);
      v16bf f;
#pragma unroll
      for (int t = 0; t < 8; t++) { f[t] = lo[t]; f[t + 8] = hi[t]; }
      bfrag[j] = f;
    }

#pragma unroll
    for (int i = 0; i < 2; i++)
#pragma unroll
      for (int j = 0; j < 4; j++)
        acc[i][j] = __builtin_amdgcn_wmma_f32_16x16x32_bf16(
            false, afrag[i], false, bfrag[j], (short)0, acc[i][j], false, false);

    __syncthreads();
  }

  // ---- epilogue: alpha, optional residual, optional ReLU ----
#pragma unroll
  for (int i = 0; i < 2; i++)
#pragma unroll
    for (int j = 0; j < 4; j++)
#pragma unroll
      for (int r = 0; r < 8; r++) {
        int m = m0 + wm * 32 + i * 16 + hi16 * 8 + r;
        int n = n0 + wn * 64 + j * 16 + lr;
        if (m < p.M && n < p.N) {
          float v = acc[i][j][r] * p.alpha;
          if (p.flags & 2) v += Rb[(i64)m * p.rm + n];
          if (p.flags & 1) v = v > 0.0f ? v : 0.0f;
          Db[(i64)m * p.dm + n] = v;
        }
      }
}

// ---------------------------------------------------------------------------
// TDM device-to-device copy: each single-wave block DMAs a 64KB chunk
// global -> LDS -> global via the Tensor Data Mover (tensor_load_to_lds /
// tensor_store_from_lds + s_wait_tensorcnt). Falls back to a plain copy when
// the TDM builtins are unavailable. nfloats must be a multiple of TDM_CHUNK.
// ---------------------------------------------------------------------------
__device__ __forceinline__ u32x4 tdm_group0(unsigned long long gaddr) {
  u32x4 g0;
  g0[0] = 1u;                                   // count=1, no gather, user mode
  g0[1] = 0u;                                   // lds_addr = base of our LDS buf
  g0[2] = (unsigned)(gaddr & 0xffffffffull);    // global_addr[31:0]
  g0[3] = (unsigned)((gaddr >> 32) & 0x1ffffffull) | 0x80000000u; // addr[56:32] | type=2
  return g0;
}
__device__ __forceinline__ i32x8 tdm_group1() {
  i32x8 g1;
  g1[0] = (2 << 16);                  // workgroup_mask=0, data_size=2 (4 bytes)
  g1[1] = (TDM_CHUNK / 4) << 16;      // tensor_dim0 low16 (=4096) in [63:48]
  g1[2] = (4 << 16);                  // tensor_dim0 hi16=0 | tensor_dim1=4
  g1[3] = (TDM_CHUNK / 4) << 16;      // tensor_dim1 hi16=0 | tile_dim0=4096
  g1[4] = 4;                          // tile_dim1=4, tile_dim2=0
  g1[5] = TDM_CHUNK / 4;              // tensor_dim0_stride low32 = 4096
  g1[6] = 0;                          // stride hi | dim1_stride lo
  g1[7] = 0;
  return g1;
}

__global__ __launch_bounds__(32) void k_tdm_copy(float* dst, const float* src, i64 chunkFloats) {
  __shared__ float buf[TDM_CHUNK];
  i64 off = (i64)blockIdx.x * chunkFloats;
#if HAVE_TDM
  (void)buf;
  u32x4 gl = tdm_group0((unsigned long long)(uintptr_t)(src + off));
  u32x4 gs = tdm_group0((unsigned long long)(uintptr_t)(dst + off));
  i32x8 g1 = tdm_group1();
  i32x4 gz = {0, 0, 0, 0};
#if __clang_major__ >= 23
  i32x8 gz8 = {0, 0, 0, 0, 0, 0, 0, 0};
  __builtin_amdgcn_tensor_load_to_lds(gl, g1, gz, gz, gz8, 0);
  __builtin_amdgcn_s_wait_tensorcnt(0);
  __builtin_amdgcn_tensor_store_from_lds(gs, g1, gz, gz, gz8, 0);
  __builtin_amdgcn_s_wait_tensorcnt(0);
#else
  __builtin_amdgcn_tensor_load_to_lds(gl, g1, gz, gz, 0);
  __builtin_amdgcn_s_wait_tensorcnt(0);
  __builtin_amdgcn_tensor_store_from_lds(gs, g1, gz, gz, 0);
  __builtin_amdgcn_s_wait_tensorcnt(0);
#endif
#else
  (void)buf;
  for (i64 i = threadIdx.x; i < chunkFloats; i += 32) dst[off + i] = src[off + i];
#endif
}

// ---------------------------------------------------------------------------
// Elementwise / reduction kernels
// ---------------------------------------------------------------------------
__global__ void k_zero(float* p, i64 n) {
  i64 i = (i64)blockIdx.x * blockDim.x + threadIdx.x;
  if (i < n) p[i] = 0.0f;
}
__global__ void k_add(float* a, const float* b, i64 n) {
  i64 i = (i64)blockIdx.x * blockDim.x + threadIdx.x;
  if (i < n) a[i] += b[i];
}

__global__ void k_rms(const float* x, const float* w, float* y, int Dd) {
  __shared__ float red[256];
  i64 row = blockIdx.x;
  const float* xr = x + row * Dd;
  float* yr = y + row * Dd;
  float s = 0.0f;
  for (int i = threadIdx.x; i < Dd; i += 256) { float v = xr[i]; s += v * v; }
  red[threadIdx.x] = s; __syncthreads();
  for (int o = 128; o > 0; o >>= 1) {
    if (threadIdx.x < o) red[threadIdx.x] += red[threadIdx.x + o];
    __syncthreads();
  }
  float r = rsqrtf(red[0] / Dd + 1e-6f);
  for (int i = threadIdx.x; i < Dd; i += 256) yr[i] = w[i] * xr[i] * r;
}

__global__ void k_softmax(float* sc, const float* bias, int S_, int H_) {
  __shared__ float red[256];
  i64 row = blockIdx.x;                 // over B*H*S rows
  int q = (int)(row % S_);
  int h = (int)((row / S_) % H_);
  float* pr = sc + row * S_;
  const float* bp = bias ? bias + ((i64)h * S_ + q) * S_ : nullptr;
  float mx = -3.0e38f;
  for (int i = threadIdx.x; i < S_; i += 256) {
    float v = pr[i] + (bp ? bp[i] : 0.0f);
    pr[i] = v; mx = fmaxf(mx, v);
  }
  red[threadIdx.x] = mx; __syncthreads();
  for (int o = 128; o > 0; o >>= 1) {
    if (threadIdx.x < o) red[threadIdx.x] = fmaxf(red[threadIdx.x], red[threadIdx.x + o]);
    __syncthreads();
  }
  mx = red[0]; __syncthreads();
  float s = 0.0f;
  for (int i = threadIdx.x; i < S_; i += 256) { float e = __expf(pr[i] - mx); pr[i] = e; s += e; }
  red[threadIdx.x] = s; __syncthreads();
  for (int o = 128; o > 0; o >>= 1) {
    if (threadIdx.x < o) red[threadIdx.x] += red[threadIdx.x + o];
    __syncthreads();
  }
  float inv = 1.0f / red[0];
  for (int i = threadIdx.x; i < S_; i += 256) pr[i] *= inv;
}

__global__ void k_posbias(const float* table, float* bias, int S_, int bidir, int causal) {
  int idx = blockIdx.x * 256 + threadIdx.x;
  if (idx >= S_ * S_) return;
  int q = idx / S_, kk = idx % S_;
  int rel = kk - q;
  int nb = 32, ret = 0, n;
  if (bidir) { nb = 16; ret = rel > 0 ? 16 : 0; n = rel < 0 ? -rel : rel; }
  else       { n = rel < 0 ? -rel : 0; }
  int me = nb / 2;
  int b;
  if (n < me) b = n;
  else {
    int lg = me + (int)(logf((float)n / (float)me) / logf(128.0f / (float)me) * (float)(nb - me));
    b = lg < nb - 1 ? lg : nb - 1;
  }
  b += ret;
  float neg = (causal && kk > q) ? NEGc : 0.0f;
  for (int h = 0; h < Hc; h++)
    bias[((i64)h * S_ + q) * S_ + kk] = table[b * Hc + h] + neg;
}

__global__ void k_sqnorm(const float* x, float* out, int Dd) {
  __shared__ float red[256];
  i64 row = blockIdx.x;
  const float* xr = x + row * Dd;
  float s = 0.0f;
  for (int i = threadIdx.x; i < Dd; i += 256) { float v = xr[i]; s += v * v; }
  red[threadIdx.x] = s; __syncthreads();
  for (int o = 128; o > 0; o >>= 1) {
    if (threadIdx.x < o) red[threadIdx.x] += red[threadIdx.x + o];
    __syncthreads();
  }
  if (threadIdx.x == 0) out[row] = red[0];
}

__global__ void k_argmin(const float* dots, const float* cbn, int* codes, int l, int Kd) {
  __shared__ float rv[256];
  __shared__ int   ri[256];
  i64 row = blockIdx.x;
  const float* dr = dots + row * Kd;
  float best = 3.0e38f; int bi = 0;
  for (int i = threadIdx.x; i < Kd; i += 256) {
    float s = cbn[i] - 2.0f * dr[i];
    if (s < best) { best = s; bi = i; }
  }
  rv[threadIdx.x] = best; ri[threadIdx.x] = bi; __syncthreads();
  for (int o = 128; o > 0; o >>= 1) {
    if (threadIdx.x < o && rv[threadIdx.x + o] < rv[threadIdx.x]) {
      rv[threadIdx.x] = rv[threadIdx.x + o]; ri[threadIdx.x] = ri[threadIdx.x + o];
    }
    __syncthreads();
  }
  if (threadIdx.x == 0) codes[row * LCc + l] = ri[0];
}

__global__ void k_res_sub(float* res, const float* cb, const int* codes, int l, int Dd, i64 n) {
  i64 i = (i64)blockIdx.x * blockDim.x + threadIdx.x;
  if (i >= n) return;
  i64 r = i / Dd; int d = (int)(i % Dd);
  res[i] -= cb[(i64)codes[r * LCc + l] * Dd + d];
}

__global__ void k_sem(const int* codes, const int* ptypes, const float* sem_emb,
                      const float* path_emb, const float* pos_emb, float* sem, int Dd, i64 n) {
  i64 i = (i64)blockIdx.x * blockDim.x + threadIdx.x;
  if (i >= n) return;
  i64 r = i / Dd; int d = (int)(i % Dd);
  int s = (int)(r % Sc);
  float acc = path_emb[(i64)ptypes[r] * Dd + d] + pos_emb[(i64)s * Dd + d];
#pragma unroll
  for (int l = 0; l < LCc; l++)
    acc += sem_emb[((i64)l * Kc + codes[r * LCc + l]) * Dd + d];
  sem[i] = acc;
}

__global__ void k_dec_embed(const int* codes, const float* emb, float* dd, int Dd, i64 n) {
  i64 i = (i64)blockIdx.x * blockDim.x + threadIdx.x;
  if (i >= n) return;
  i64 r = i / Dd; int d = (int)(i % Dd);
  int s = (int)(r % Sc);
  int tok = (s == 0) ? 0 : codes[(r - 1) * LCc];
  dd[i] = emb[(i64)tok * Dd + d];
}

__global__ void k_gate(const float* x, const float* gw, float* wfull, int rows) {
  int r = blockIdx.x * blockDim.x + threadIdx.x;
  if (r >= rows) return;
  float lg[Ec];
#pragma unroll
  for (int e = 0; e < Ec; e++) {
    float s = 0.0f;
    for (int d = 0; d < Dc; d++) s += x[(i64)r * Dc + d] * gw[(i64)d * Ec + e];
    lg[e] = s;
  }
  float mx = lg[0];
  for (int e = 1; e < Ec; e++) mx = fmaxf(mx, lg[e]);
  float sum = 0.0f;
  for (int e = 0; e < Ec; e++) { lg[e] = __expf(lg[e] - mx); sum += lg[e]; }
  for (int e = 0; e < Ec; e++) lg[e] /= sum;
  int i1 = 0; float v1 = lg[0];
  for (int e = 1; e < Ec; e++) if (lg[e] > v1) { v1 = lg[e]; i1 = e; }
  int i2 = -1; float v2 = -1.0f;
  for (int e = 0; e < Ec; e++) if (e != i1 && lg[e] > v2) { v2 = lg[e]; i2 = e; }
  float ws = v1 + v2;
  for (int e = 0; e < Ec; e++)
    wfull[(i64)r * Ec + e] = (e == i1) ? v1 / ws : (e == i2) ? v2 / ws : 0.0f;
}

__global__ void k_combine(float* acc, const float* y, const float* wf, int e, int Dd, i64 n) {
  i64 i = (i64)blockIdx.x * blockDim.x + threadIdx.x;
  if (i >= n) return;
  i64 r = i / Dd;
  acc[i] += y[i] * wf[r * Ec + e];
}

__global__ void k_loss(const float* logits, const int* codes, float* loss, int Kd, float invN) {
  __shared__ float red[256];
  i64 row = blockIdx.x;
  const float* lr = logits + row * Kd;
  float mx = -3.0e38f;
  for (int i = threadIdx.x; i < Kd; i += 256) mx = fmaxf(mx, lr[i]);
  red[threadIdx.x] = mx; __syncthreads();
  for (int o = 128; o > 0; o >>= 1) {
    if (threadIdx.x < o) red[threadIdx.x] = fmaxf(red[threadIdx.x], red[threadIdx.x + o]);
    __syncthreads();
  }
  mx = red[0]; __syncthreads();
  float s = 0.0f;
  for (int i = threadIdx.x; i < Kd; i += 256) s += __expf(lr[i] - mx);
  red[threadIdx.x] = s; __syncthreads();
  for (int o = 128; o > 0; o >>= 1) {
    if (threadIdx.x < o) red[threadIdx.x] += red[threadIdx.x + o];
    __syncthreads();
  }
  if (threadIdx.x == 0) {
    int label = codes[row * LCc];
    float lp = lr[label] - mx - logf(red[0]);
    atomicAdd(loss, -lp * invN);
  }
}

__global__ void k_writeout(float* out, const int* codes, const float* sem, i64 rows, int Dd, i64 outMax) {
  i64 i = (i64)blockIdx.x * blockDim.x + threadIdx.x;
  i64 total = rows + rows * LCc + rows * Dd;
  if (i >= total) return;
  i64 o;
  float v;
  if (i < rows)                 { o = 1 + i;                     v = (float)codes[i * LCc]; }
  else if (i < rows * (LCc+1))  { i64 j = i - rows;              o = 1 + rows + j;          v = (float)codes[j]; }
  else                          { i64 j = i - rows * (LCc + 1);  o = 1 + rows * (LCc+1) + j; v = sem[j]; }
  if (o < outMax) out[o] = v;
}

// ---------------------------------------------------------------------------
// Host-side orchestration
// ---------------------------------------------------------------------------
static void gemm(hipStream_t st,
                 const float* A, i64 am, i64 ak, i64 ab, i64 ah,
                 const float* B, i64 bk, i64 bn, i64 bb, i64 bh,
                 float* D, i64 dm, i64 db, i64 dh,
                 int M, int N, int K, int batches, int zdiv,
                 float alpha, int flags,
                 const float* R = nullptr, i64 rm = 0, i64 rb = 0, i64 rh = 0) {
  GemmP p{A, am, ak, ab, ah, B, bk, bn, bb, bh, R, rm, rb, rh, D, dm, db, dh,
          M, N, K, zdiv, flags, alpha};
  dim3 g((N + BN - 1) / BN, (M + BM - 1) / BM, batches);
  k_gemm<<<g, 256, 0, st>>>(p);
}

static void tdm_copy(hipStream_t st, float* dst, const float* src, i64 nfloats) {
  int nchunks = (int)(nfloats / TDM_CHUNK);
  if (nchunks > 0)
    k_tdm_copy<<<nchunks, 32, 0, st>>>(dst, src, (i64)TDM_CHUNK);
}

static void attention(hipStream_t st, const float* xq, const float* xkv,
                      const float* wq, const float* wk, const float* wv, const float* wo,
                      const float* bias, float* hres,
                      float* qb, float* kb, float* vb, float* sc, float* ao) {
  const i64 HD = (i64)Hc * DKc;   // 512
  gemm(st, xq,  Dc, 1, 0, 0, wq, HD, 1, 0, 0, qb, HD, 0, 0, Rw, (int)HD, Dc, 1, 1, 1.0f, 0);
  gemm(st, xkv, Dc, 1, 0, 0, wk, HD, 1, 0, 0, kb, HD, 0, 0, Rw, (int)HD, Dc, 1, 1, 1.0f, 0);
  gemm(st, xkv, Dc, 1, 0, 0, wv, HD, 1, 0, 0, vb, HD, 0, 0, Rw, (int)HD, Dc, 1, 1, 1.0f, 0);
  // scores[b,h,q,k] = q . k  (batched over b,h with per-head strides)
  gemm(st, qb, HD, 1, (i64)Sc * HD, DKc,
       kb, 1, HD, (i64)Sc * HD, DKc,
       sc, Sc, (i64)Hc * Sc * Sc, (i64)Sc * Sc,
       Sc, Sc, DKc, Bc * Hc, Hc, 1.0f, 0);
  k_softmax<<<Bc * Hc * Sc, 256, 0, st>>>(sc, bias, Sc, Hc);
  // o[b,q,h,d] = sum_k a[b,h,q,k] v[b,k,h,d]
  gemm(st, sc, Sc, 1, (i64)Hc * Sc * Sc, (i64)Sc * Sc,
       vb, HD, 1, (i64)Sc * HD, DKc,
       ao, HD, (i64)Sc * HD, DKc,
       Sc, DKc, Sc, Bc * Hc, Hc, 1.0f, 0);
  // out proj + residual in place
  gemm(st, ao, HD, 1, 0, 0, wo, Dc, 1, 0, 0, hres, Dc, 0, 0,
       Rw, Dc, (int)HD, 1, 1, 1.0f, 2, hres, Dc, 0, 0);
}

extern "C" void kernel_launch(void* const* d_in, const int* in_sizes, int n_in,
                              void* d_out, int out_size, void* d_ws, size_t ws_size,
                              hipStream_t stream) {
  (void)in_sizes; (void)n_in; (void)ws_size;
  const float* tokens     = (const float*)d_in[0];
  const int*   ptypes     = (const int*)  d_in[1];
  const float* codebooks  = (const float*)d_in[2];
  const float* sem_emb    = (const float*)d_in[3];
  const float* path_emb   = (const float*)d_in[4];
  const float* pos_emb    = (const float*)d_in[5];
  const float* shared_emb = (const float*)d_in[6];
  const float* enc_rel    = (const float*)d_in[7];
  const float* dec_rel    = (const float*)d_in[8];
  const float* enc_fln    = (const float*)d_in[9];
  const float* dec_fln    = (const float*)d_in[10];
  auto encL = [&](int l, int t) { return (const float*)d_in[11 + l * 8 + t]; };
  auto decL = [&](int l, int t) { return (const float*)d_in[11 + NL * 8 + l * 14 + t]; };

  // bump allocator over workspace
  char* base = (char*)d_ws; size_t off = 0;
  auto alloc = [&](size_t nfloats) -> float* {
    float* p = (float*)(base + off);
    off += ((nfloats * sizeof(float) + 255) / 256) * 256;
    return p;
  };
  float* res    = alloc((i64)Rw * Dc);
  float* dots   = alloc((i64)Rw * Kc);            // also reused for logits
  float* cbn    = alloc(Kc);
  int*   codes  = (int*)alloc((i64)Rw * LCc);
  float* sem    = alloc((i64)Rw * Dc);
  float* h      = alloc((i64)Rw * Dc);
  float* xn     = alloc((i64)Rw * Dc);
  float* qb     = alloc((i64)Rw * Dc);
  float* kb     = alloc((i64)Rw * Dc);
  float* vb     = alloc((i64)Rw * Dc);
  float* sc     = alloc((i64)Bc * Hc * Sc * Sc);
  float* ao     = alloc((i64)Rw * Dc);
  float* ff     = alloc((i64)Rw * DFFc);
  float* ye     = alloc((i64)Rw * Dc);
  float* macc   = alloc((i64)Rw * Dc);
  float* encout = alloc((i64)Rw * Dc);
  float* dstate = alloc((i64)Rw * Dc);
  float* bias_e = alloc((i64)Hc * Sc * Sc);
  float* bias_d = alloc((i64)Hc * Sc * Sc);
  float* wfull  = alloc((i64)Rw * Ec);

  const i64 nRD  = (i64)Rw * Dc;
  const int gRD  = (int)((nRD + 255) / 256);

  // ---- residual VQ quantize ----
  tdm_copy(stream, res, tokens, nRD);   // Tensor Data Mover D2D bounce
  for (int l = 0; l < LCc; l++) {
    const float* cb = codebooks + (i64)l * Kc * Dc;
    k_sqnorm<<<Kc, 256, 0, stream>>>(cb, cbn, Dc);
    gemm(stream, res, Dc, 1, 0, 0, cb, 1, Dc, 0, 0, dots, Kc, 0, 0,
         Rw, Kc, Dc, 1, 1, 1.0f, 0);
    k_argmin<<<Rw, 256, 0, stream>>>(dots, cbn, codes, l, Kc);
    k_res_sub<<<gRD, 256, 0, stream>>>(res, cb, codes, l, Dc, nRD);
  }

  // ---- embeddings + positional relative biases ----
  k_sem<<<gRD, 256, 0, stream>>>(codes, ptypes, sem_emb, path_emb, pos_emb, sem, Dc, nRD);
  tdm_copy(stream, h, sem, nRD);        // Tensor Data Mover D2D bounce
  int gSS = (Sc * Sc + 255) / 256;
  k_posbias<<<gSS, 256, 0, stream>>>(enc_rel, bias_e, Sc, 1, 0);
  k_posbias<<<gSS, 256, 0, stream>>>(dec_rel, bias_d, Sc, 0, 1);

  // ---- encoder ----
  for (int l = 0; l < NL; l++) {
    k_rms<<<Rw, 256, 0, stream>>>(h, encL(l, 0), xn, Dc);
    attention(stream, xn, xn, encL(l, 4), encL(l, 5), encL(l, 6), encL(l, 7),
              bias_e, h, qb, kb, vb, sc, ao);
    k_rms<<<Rw, 256, 0, stream>>>(h, encL(l, 1), xn, Dc);
    gemm(stream, xn, Dc, 1, 0, 0, encL(l, 2), DFFc, 1, 0, 0, ff, DFFc, 0, 0,
         Rw, DFFc, Dc, 1, 1, 1.0f, 1);                                     // ReLU
    gemm(stream, ff, DFFc, 1, 0, 0, encL(l, 3), Dc, 1, 0, 0, h, Dc, 0, 0,
         Rw, Dc, DFFc, 1, 1, 1.0f, 2, h, Dc, 0, 0);                        // +residual
  }
  k_rms<<<Rw, 256, 0, stream>>>(h, enc_fln, encout, Dc);

  // ---- decoder ----
  k_dec_embed<<<gRD, 256, 0, stream>>>(codes, shared_emb, dstate, Dc, nRD);
  for (int l = 0; l < NL; l++) {
    k_rms<<<Rw, 256, 0, stream>>>(dstate, decL(l, 0), xn, Dc);
    attention(stream, xn, xn, decL(l, 6), decL(l, 7), decL(l, 8), decL(l, 9),
              bias_d, dstate, qb, kb, vb, sc, ao);
    k_rms<<<Rw, 256, 0, stream>>>(dstate, decL(l, 1), xn, Dc);
    attention(stream, xn, encout, decL(l, 10), decL(l, 11), decL(l, 12), decL(l, 13),
              nullptr, dstate, qb, kb, vb, sc, ao);
    k_rms<<<Rw, 256, 0, stream>>>(dstate, decL(l, 2), xn, Dc);
    // dense MoE over all experts with top-2 combine weights
    k_gate<<<(Rw + 255) / 256, 256, 0, stream>>>(xn, decL(l, 3), wfull, Rw);
    k_zero<<<gRD, 256, 0, stream>>>(macc, nRD);
    for (int e = 0; e < Ec; e++) {
      const float* ewi = decL(l, 4) + (i64)e * Dc * DFFc;
      const float* ewo = decL(l, 5) + (i64)e * DFFc * Dc;
      gemm(stream, xn, Dc, 1, 0, 0, ewi, DFFc, 1, 0, 0, ff, DFFc, 0, 0,
           Rw, DFFc, Dc, 1, 1, 1.0f, 1);
      gemm(stream, ff, DFFc, 1, 0, 0, ewo, Dc, 1, 0, 0, ye, Dc, 0, 0,
           Rw, Dc, DFFc, 1, 1, 1.0f, 0);
      k_combine<<<gRD, 256, 0, stream>>>(macc, ye, wfull, e, Dc, nRD);
    }
    k_add<<<gRD, 256, 0, stream>>>(dstate, macc, nRD);
  }

  // ---- logits + loss ----
  k_rms<<<Rw, 256, 0, stream>>>(dstate, dec_fln, xn, Dc);
  float invsqrtD = 1.0f / sqrtf((float)Dc);
  gemm(stream, xn, Dc, 1, 0, 0, shared_emb, 1, Dc, 0, 0, dots, Kc, 0, 0,
       Rw, Kc, Dc, 1, 1, invsqrtD, 0);                 // logits in `dots`
  k_zero<<<1, 256, 0, stream>>>((float*)d_out, 1);
  k_loss<<<Rw, 256, 0, stream>>>(dots, codes, (float*)d_out, Kc, 1.0f / (float)Rw);

  // ---- outputs: (loss, labels, codes, sem) flattened ----
  i64 wtotal = (i64)Rw + (i64)Rw * LCc + nRD;
  k_writeout<<<(int)((wtotal + 255) / 256), 256, 0, stream>>>(
      (float*)d_out, codes, sem, Rw, Dc, (i64)out_size);
}
